// STMetaLSTM_11828339933709
// MI455X (gfx1250) — matrix operand order repaired and checked by reading, not compiled, not run on hardware
//
#include <hip/hip_runtime.h>
#include <hip/hip_bf16.h>
#include <stdint.h>

// ---------------------------------------------------------------------------
// STMetaLSTM for MI455X (gfx1250, wave32).
//
// Roofline: dominant cost is the shared-weight GEMM Hh(6624x64)@wh2(64x16384)
// (13.9 GFLOP) -> f16 WMMA (v_wmma_f32_16x16x32_f16). The LSTM recurrence is
// a per-pair batched matvec (no shared operand -> not WMMA-able); it is
// Wh-bandwidth bound, so Wh is produced once to workspace in f16 (217MB) and
// each workgroup keeps its pair's 32KB Wh tile resident in LDS for all 12
// timesteps (Wh crosses HBM once each way: ~19us at 23.3 TB/s). The 32KB
// LDS staging is done by the Tensor Data Mover (tensor_load_to_lds).
//
// Workspace layout (requires ws_size >= ~232 MB):
//   Hh  f16 [6624 x 64]      @ 0
//   Wx  f32 [6624 x 256]     @ 847872
//   Bv  f32 [6624 x 256]     @ 7630848
//   Wh  f16 [6624 x 16384]   @ 14413824
// ---------------------------------------------------------------------------

#define B_   32
#define T_   12
#define N_   207
#define H_   64
#define LH_  64
#define G4_  256          // 4*H
#define WHN_ 16384        // 4*H*H
#define OUTS_ 12
#define TOD_ 24
#define DOW_ 7
#define BN_  (B_*N_)      // 6624 = 16*414 = 32*207

#define HH_OFF 0
#define WX_OFF 847872
#define BV_OFF 7630848
#define WH_OFF 14413824

#define LDS_PITCH 72      // padded half-stride: 144B = 9*16B -> 16B aligned,
                          // 36-dword lane stride -> conflict-free b128 reads

typedef __attribute__((ext_vector_type(16))) _Float16 v16h;
typedef __attribute__((ext_vector_type(8)))  float    v8f;
typedef __attribute__((ext_vector_type(4)))  unsigned int u32x4;
typedef __attribute__((ext_vector_type(8)))  int      i32x8;
typedef __attribute__((ext_vector_type(4)))  int      i32x4;

#if __has_builtin(__builtin_amdgcn_tensor_load_to_lds) && \
    __has_builtin(__builtin_amdgcn_s_wait_tensorcnt)
#define HAVE_TDM 1
#endif

union FragH { v16h h; uint4 q[2]; };

__device__ __forceinline__ float sigmoidf_(float v) {
    return 1.0f / (1.0f + __expf(-v));
}

// ---------------------------------------------------------------------------
// Kernel 1: per-(b,n) meta MLP layer-1 (one-hot folded to row adds) for the
// three MLPs, plus the small layer-2 outputs Wx (256) and b (256).
// Writes Hh in f16 as the A operand of the big WMMA GEMM.
// ---------------------------------------------------------------------------
__global__ __launch_bounds__(256) void meta_mlp_kernel(
    const float* __restrict__ x,     const float* __restrict__ node_emb,
    const float* __restrict__ wx1_w, const float* __restrict__ wx1_b,
    const float* __restrict__ wx2_w, const float* __restrict__ wx2_b,
    const float* __restrict__ wh1_w, const float* __restrict__ wh1_b,
    const float* __restrict__ b1_w,  const float* __restrict__ b1_b,
    const float* __restrict__ b2_w,  const float* __restrict__ b2_b,
    _Float16* __restrict__ Hh, float* __restrict__ Wx, float* __restrict__ Bv)
{
    const int bn  = blockIdx.x;
    const int b   = bn / N_;
    const int n   = bn % N_;
    const int tid = threadIdx.x;

    __shared__ float node[64];
    __shared__ float hid[3][64];   // [0]=wx path, [1]=wh path, [2]=b path

    // last-timestep tod/dow indices (meta = [node(64), tod_1hot(24), dow_1hot(7)])
    const int xb = ((b * T_ + (T_ - 1)) * N_ + n) * 3;
    int ti = (int)(x[xb + 1] * (float)TOD_);
    ti = ti < 0 ? 0 : (ti > TOD_ - 1 ? TOD_ - 1 : ti);
    int di = (int)x[xb + 2];
    di = di < 0 ? 0 : (di > DOW_ - 1 ? DOW_ - 1 : di);

    if (tid < 64) node[tid] = node_emb[n * 64 + tid];
    __syncthreads();

    if (tid < 192) {
        const int m = tid >> 6, j = tid & 63;
        const float* w1 = (m == 0) ? wx1_w : (m == 1) ? wh1_w : b1_w;
        const float* b1 = (m == 0) ? wx1_b : (m == 1) ? wh1_b : b1_b;
        // one-hot parts of meta become plain row adds
        float acc = b1[j] + w1[(64 + ti) * LH_ + j] + w1[(88 + di) * LH_ + j];
        #pragma unroll 8
        for (int k = 0; k < 64; ++k)
            acc = fmaf(node[k], w1[k * LH_ + j], acc);
        acc = fmaxf(acc, 0.0f);                 // relu
        hid[m][j] = acc;
        if (m == 1) Hh[bn * 64 + j] = (_Float16)acc;   // GEMM A operand (f16)
    }
    __syncthreads();

    // layer-2 for Wx and b (tiny: 64x256 each) — thread d owns gate-dim d
    {
        const int d = tid;
        float accx = wx2_b[d];
        float accb = b2_b[d];
        #pragma unroll 8
        for (int j = 0; j < 64; ++j) {
            accx = fmaf(hid[0][j], wx2_w[j * G4_ + d], accx);
            accb = fmaf(hid[2][j], b2_w [j * G4_ + d], accb);
        }
        Wx[bn * G4_ + d] = accx;
        Bv[bn * G4_ + d] = accb;
    }
}

// ---------------------------------------------------------------------------
// Kernel 2: Wh = Hh(6624x64) @ wh2(64x16384) + wh2_b, f16 in / f32 acc /
// f16 out, via v_wmma_f32_16x16x32_f16.
// Workgroup = 256 thr (8 waves) computes a 32(M) x 256(N) tile.
// Wave w: rows (w>>2)*16, cols (w&3)*64 -> 4 N-subtiles x 2 K-steps = 8 WMMA.
// LDS layouts are arranged so every WMMA fragment is two 16B-aligned,
// bank-conflict-free ds_load_b128s (B stored transposed, both tiles padded
// to a 72-half pitch).
// ---------------------------------------------------------------------------
__global__ __launch_bounds__(256) void wh_gemm_kernel(
    const _Float16* __restrict__ Hh, const float* __restrict__ wh2_w,
    const float* __restrict__ wh2_b, _Float16* __restrict__ Wh)
{
    const int n0   = blockIdx.x * 256;   // 64 N-tiles
    const int m0   = blockIdx.y * 32;    // 207 M-tiles
    const int tid  = threadIdx.x;
    const int lane = tid & 31;
    const int wave = tid >> 5;
    const int hi   = lane >> 4;          // half-wave select
    const int lo   = lane & 15;
    const int wm   = (wave >> 2) * 16;   // 0 or 16
    const int wn   = (wave & 3) * 64;    // 0,64,128,192

    __shared__ __align__(16) _Float16 BsT[256 * LDS_PITCH]; // wh2 tile, transposed [n][k]
    __shared__ __align__(16) _Float16 As [32  * LDS_PITCH]; // Hh tile, [m][k]

    // stage B tile transposed (coalesced float4 global reads over n)
    for (int i = tid * 4; i < 64 * 256; i += 256 * 4) {
        const int k = i >> 8, j = i & 255;
        const float4 f = *(const float4*)(wh2_w + (size_t)k * WHN_ + n0 + j);
        BsT[(j + 0) * LDS_PITCH + k] = (_Float16)f.x;
        BsT[(j + 1) * LDS_PITCH + k] = (_Float16)f.y;
        BsT[(j + 2) * LDS_PITCH + k] = (_Float16)f.z;
        BsT[(j + 3) * LDS_PITCH + k] = (_Float16)f.w;
    }
    // stage A tile
    for (int i = tid; i < 32 * 64; i += 256)
        As[(i >> 6) * LDS_PITCH + (i & 63)] = Hh[(m0 + (i >> 6)) * 64 + (i & 63)];
    __syncthreads();

    v8f acc[4] = {};
    #pragma unroll
    for (int kk = 0; kk < 2; ++kk) {               // K = 64 in two 32-steps
        // A fragment 16x32 f16: element e -> row M=lane&15,
        // k = kk*32 + 16*hi + e  => one contiguous 32B run
        FragH a;
        const _Float16* arow = &As[(wm + lo) * LDS_PITCH + kk * 32 + 16 * hi];
        a.q[0] = *(const uint4*)(arow);        // e=0..7
        a.q[1] = *(const uint4*)(arow + 8);    // e=8..15
        #pragma unroll
        for (int s = 0; s < 4; ++s) {          // four 16-wide N subtiles
            // B fragment 32x16 f16: col N=lane&15, k = kk*32 + 16*hi + e
            // => one contiguous 32B run in the transposed tile
            FragH bf;
            const _Float16* bcol =
                &BsT[(wn + s * 16 + lo) * LDS_PITCH + kk * 32 + 16 * hi];
            bf.q[0] = *(const uint4*)(bcol);       // e=0..7
            bf.q[1] = *(const uint4*)(bcol + 8);   // e=8..15
            acc[s] = __builtin_amdgcn_wmma_f32_16x16x32_f16(
                false, a.h, false, bf.h, (short)0, acc[s], false, false);
        }
    }

    // C/D layout: element v -> row M = v + 8*hi, col N = lane&15
    #pragma unroll
    for (int s = 0; s < 4; ++s) {
        #pragma unroll
        for (int v = 0; v < 8; ++v) {
            const int row = m0 + wm + v + 8 * hi;
            const int col = n0 + wn + s * 16 + lo;
            Wh[(size_t)row * WHN_ + col] = (_Float16)(acc[s][v] + wh2_b[col]);
        }
    }
}

// ---------------------------------------------------------------------------
// Kernel 3: one workgroup per (b,n). Stage this pair's Wh (32 KB f16) into
// LDS once — via the Tensor Data Mover (1-D D# descriptor, data_size=2B,
// tile_dim0=16384) — then run the 12 sequential LSTM steps: thread d owns
// gate dim d (gates = xt*Wx + h@Wh + b), threads 0..63 apply the cell
// nonlinearity. Finishes with the 64->12 decoder.
// ---------------------------------------------------------------------------
__global__ __launch_bounds__(256) void lstm_kernel(
    const float* __restrict__ x, const _Float16* __restrict__ Wh,
    const float* __restrict__ Wx, const float* __restrict__ Bv,
    const float* __restrict__ dec_w, const float* __restrict__ dec_b,
    float* __restrict__ out)
{
    const int bn  = blockIdx.x;
    const int b   = bn / N_;
    const int n   = bn % N_;
    const int tid = threadIdx.x;

    __shared__ uint4 whs4[WHN_ / 8];          // 32 KB, 16B-aligned
    __shared__ float gates[G4_];
    __shared__ float hbuf[H_], cbuf[H_];
    __shared__ float xts[T_];
    _Float16* whs = (_Float16*)whs4;

#ifdef HAVE_TDM
    if (tid < 32) {  // wave 0 issues one TDM descriptor for the 32KB tile
        const uint64_t gaddr = (uint64_t)(uintptr_t)(Wh + (size_t)bn * WHN_);
        const unsigned lds_addr = (unsigned)(uintptr_t)&whs4[0];
        // D# group 0: count=1 (valid), lds_addr, 57-bit global addr, type=2
        u32x4 g0 = { 1u,
                     lds_addr,
                     (unsigned)(gaddr & 0xFFFFFFFFu),
                     (unsigned)((gaddr >> 32) & 0x01FFFFFFu) | 0x80000000u };
        // D# group 1: data_size=1(2B) @bits[17:16]; tensor_dim0=16384
        // @bits[79:48]; tensor_dim1=1 @bits[111:80]; tile_dim0=16384
        // @bits[127:112]; tile_dim1/2=0 (1-D); tensor_dim0_stride=16384
        // @bits[207:160]; tensor_dim1_stride=0.
        i32x8 g1 = { 1 << 16,
                     (int)(16384u << 16),
                     1 << 16,
                     (int)(16384u << 16),
                     0,
                     16384,
                     0,
                     0 };
        i32x4 gz4 = { 0, 0, 0, 0 };
        i32x8 gz8 = { 0, 0, 0, 0, 0, 0, 0, 0 };
        // clang-23 / therock lane: 6-arg form (g0, g1, g2, g3, ext, cpol)
        __builtin_amdgcn_tensor_load_to_lds(g0, g1, gz4, gz4, gz8, 0);
        __builtin_amdgcn_s_wait_tensorcnt(0);
    }
#else
    {
        const uint4* src = (const uint4*)(Wh + (size_t)bn * WHN_);
        for (int i = tid; i < WHN_ / 8; i += 256) whs4[i] = src[i];
    }
#endif
    if (tid < T_) xts[tid] = x[((b * T_ + tid) * N_ + n) * 3];  // data channel
    if (tid < H_) { hbuf[tid] = 0.0f; cbuf[tid] = 0.0f; }

    const float wxv = Wx[bn * G4_ + tid];
    const float bvv = Bv[bn * G4_ + tid];
    __syncthreads();

    for (int t = 0; t < T_; ++t) {
        float acc = fmaf(xts[t], wxv, bvv);
        #pragma unroll 8
        for (int h = 0; h < H_; ++h)
            acc = fmaf(hbuf[h], (float)whs[h * G4_ + tid], acc);
        gates[tid] = acc;
        __syncthreads();
        if (tid < H_) {
            const float g  = gates[tid];
            const float ig = gates[64  + tid];
            const float fg = gates[128 + tid];
            const float og = gates[192 + tid];
            const float c  = tanhf(g) * sigmoidf_(ig) + cbuf[tid] * sigmoidf_(fg);
            cbuf[tid] = c;
            hbuf[tid] = tanhf(c) * sigmoidf_(og);
        }
        __syncthreads();
    }

    // decoder: out[b, t, n, 0] = h @ dec_w[:, t] + dec_b[t]
    if (tid < OUTS_) {
        float acc = dec_b[tid];
        #pragma unroll 8
        for (int h = 0; h < H_; ++h)
            acc = fmaf(hbuf[h], dec_w[h * OUTS_ + tid], acc);
        out[((size_t)b * OUTS_ + tid) * N_ + n] = acc;
    }
}

// ---------------------------------------------------------------------------
extern "C" void kernel_launch(void* const* d_in, const int* in_sizes, int n_in,
                              void* d_out, int out_size, void* d_ws, size_t ws_size,
                              hipStream_t stream) {
    const float* x        = (const float*)d_in[0];
    const float* node_emb = (const float*)d_in[1];
    const float* wx1_w    = (const float*)d_in[2];
    const float* wx1_b    = (const float*)d_in[3];
    const float* wx2_w    = (const float*)d_in[4];
    const float* wx2_b    = (const float*)d_in[5];
    const float* wh1_w    = (const float*)d_in[6];
    const float* wh1_b    = (const float*)d_in[7];
    const float* wh2_w    = (const float*)d_in[8];
    const float* wh2_b    = (const float*)d_in[9];
    const float* b1_w     = (const float*)d_in[10];
    const float* b1_b     = (const float*)d_in[11];
    const float* b2_w     = (const float*)d_in[12];
    const float* b2_b     = (const float*)d_in[13];
    const float* dec_w    = (const float*)d_in[14];
    const float* dec_b    = (const float*)d_in[15];

    char* ws = (char*)d_ws;
    _Float16* Hh = (_Float16*)(ws + HH_OFF);
    float*    Wx = (float*)   (ws + WX_OFF);
    float*    Bv = (float*)   (ws + BV_OFF);
    _Float16* Wh = (_Float16*)(ws + WH_OFF);

    meta_mlp_kernel<<<BN_, 256, 0, stream>>>(
        x, node_emb, wx1_w, wx1_b, wx2_w, wx2_b,
        wh1_w, wh1_b, b1_w, b1_b, b2_w, b2_b, Hh, Wx, Bv);

    wh_gemm_kernel<<<dim3(WHN_ / 256, BN_ / 32), 256, 0, stream>>>(
        Hh, wh2_w, wh2_b, Wh);

    lstm_kernel<<<BN_, 256, 0, stream>>>(
        x, Wh, Wx, Bv, dec_w, dec_b, (float*)d_out);
}
